// NeRF_27144193311310
// MI455X (gfx1250) — matrix operand order repaired and checked by reading, not compiled, not run on hardware
//
#include <hip/hip_runtime.h>

typedef __attribute__((ext_vector_type(16))) _Float16 v16h;
typedef __attribute__((ext_vector_type(8)))  _Float16 v8h;
typedef __attribute__((ext_vector_type(8)))  float    v8f;

#define RPB 32      // rays per block
#define NS  64      // samples per ray
#define HID 128     // hidden width
#define WAVES 8

__device__ __forceinline__ v8f wmma_f16(v16h a, v16h b, v8f c) {
  // D = A(16x32 f16) * B(32x16 f16) + C(16x16 f32)
  return __builtin_amdgcn_wmma_f32_16x16x32_f16(
      /*neg_a=*/false, a, /*neg_b=*/false, b,
      /*c_mod=*/(short)0, c, /*reuse_a=*/false, /*reuse_b=*/false);
}

__global__ __launch_bounds__(256) void nerf_fused_kernel(
    const float* __restrict__ origins, const float* __restrict__ dirs,
    const float* __restrict__ W1, const float* __restrict__ b1,
    const float* __restrict__ W2, const float* __restrict__ b2,
    const float* __restrict__ W3, const float* __restrict__ b3,
    const int* __restrict__ nearp, const int* __restrict__ farp,
    const int* __restrict__ nsp, float* __restrict__ out)
{
  // W2 packed per B-fragment layout: [nt][kstep][lane][16 halves]
  __shared__ alignas(32) _Float16 w2p[HID * HID];
  // Per-wave A staging in A-fragment layout: [wave][kstep][lane][16 halves]
  __shared__ alignas(32) _Float16 apk[WAVES * 4 * 32 * 16];
  // Raw field outputs: [ray][sample][4]
  __shared__ float fbuf[RPB * NS * 4];
  __shared__ float oL[RPB * 3], dL[RPB * 3];
  __shared__ float W1s[3 * HID], b1s[HID], b2s[HID], W3s[HID * 4], b3s[4];

  const int tid  = threadIdx.x;
  const int lane = tid & 31;
  const int wv   = tid >> 5;
  const int r0   = blockIdx.x * RPB;

  // ---- stage & pack W2: 16 consecutive K for one N = 16 consecutive halves
  //      of one packed lane slot -> one 32B LDS store. Global reads coalesced
  //      across tid (N = tid&127 consecutive).
  for (int e = tid; e < (HID * HID) / 16; e += 256) {
    const int N = e & 127, Kg = e >> 7;        // K = Kg*16 + q
    v16h tmp;
    #pragma unroll
    for (int q = 0; q < 16; ++q)
      tmp[q] = (_Float16)W2[(Kg * 16 + q) * HID + N];
    const int nt = N >> 4;
    const int ks = Kg >> 1;                     // (Kg*16) >> 5
    const int ln = ((Kg & 1) << 4) | (N & 15);  // lanes 16-31 hold K%32 = 16..31
    *(v16h*)&w2p[(((nt << 2) | ks) * 32 + ln) * 16] = tmp;
  }
  if (tid < RPB * 3) { oL[tid] = origins[r0 * 3 + tid]; dL[tid] = dirs[r0 * 3 + tid]; }
  if (tid < 3 * HID) W1s[tid] = W1[tid];
  if (tid < HID) { b1s[tid] = b1[tid]; b2s[tid] = b2[tid]; }
  for (int i = tid; i < HID * 4; i += 256) W3s[i] = W3[i];
  if (tid < 4) b3s[tid] = b3[tid];

  const float nearf = (float)(*nearp);
  const float farf  = (float)(*farp);
  const float delta = (farf - nearf) / (float)(*nsp);

  __syncthreads();

  // ray row this lane feeds (A-matrix: lanes l and l+16 both cover M = l&15)
  const int m     = lane & 15;
  const int kbase = (lane >> 4) * 64;   // each half-wave covers 64 hidden channels
  // both ray groups this lane may serve (rh = 0 -> ray m, rh = 1 -> ray 16+m)
  const float ox0 = oL[m * 3 + 0], oy0 = oL[m * 3 + 1], oz0 = oL[m * 3 + 2];
  const float dx0 = dL[m * 3 + 0], dy0 = dL[m * 3 + 1], dz0 = dL[m * 3 + 2];
  const float ox1 = oL[(16 + m) * 3 + 0], oy1 = oL[(16 + m) * 3 + 1], oz1 = oL[(16 + m) * 3 + 2];
  const float dx1 = dL[(16 + m) * 3 + 0], dy1 = dL[(16 + m) * 3 + 1], dz1 = dL[(16 + m) * 3 + 2];

  _Float16* ap = &apk[wv * (4 * 32 * 16)];

  // 32 rays x 64 samples = 128 tiles of (16 rays x 1 sample); 16 tiles per wave
  for (int tt = 0; tt < 16; ++tt) {
    const int rh = tt & 1;               // ray half (rows = rh*16 .. rh*16+15)
    const int s  = wv * 8 + (tt >> 1);   // sample index for this tile
    const float ts = nearf + ((float)s + 0.5f) * delta; // depth midpoint
    const float px = rh ? fmaf(ts, dx1, ox1) : fmaf(ts, dx0, ox0);
    const float py = rh ? fmaf(ts, dy1, oy1) : fmaf(ts, dy0, oy0);
    const float pz = rh ? fmaf(ts, dz1, oz1) : fmaf(ts, dz0, oz0);

    // ---- layer 1 (3->128): 8 consecutive K = 8 consecutive halves of one
    //      A-fragment lane slot -> one ds_store_b128 per group ----
    #pragma unroll
    for (int g = 0; g < 8; ++g) {
      const int k0 = kbase + g * 8;
      v8h hv;
      #pragma unroll
      for (int q = 0; q < 8; ++q) {
        const int k = k0 + q;
        float h = b1s[k];
        h = fmaf(px, W1s[k], h);
        h = fmaf(py, W1s[HID + k], h);
        h = fmaf(pz, W1s[2 * HID + k], h);
        hv[q] = (_Float16)fmaxf(h, 0.0f);
      }
      // A 16x32 f16 layout (contiguous, matches W2 packing): K%32 in 0-15 -> lanes 0-15 h=K%32;
      // 16-31 -> lanes 16-31 h=K%32-16
      const int ks  = k0 >> 5, r = k0 & 31;
      const int lh  = (r >> 4) & 1;
      const int hh0 = r & 8;
      *(v8h*)&ap[(ks * 32 + (lh << 4) + m) * 16 + hh0] = hv;
    }
    asm volatile("s_wait_dscnt 0" ::: "memory");  // A staging RAW within wave

    const v16h af0 = *(const v16h*)&ap[(0 * 32 + lane) * 16];
    const v16h af1 = *(const v16h*)&ap[(1 * 32 + lane) * 16];
    const v16h af2 = *(const v16h*)&ap[(2 * 32 + lane) * 16];
    const v16h af3 = *(const v16h*)&ap[(3 * 32 + lane) * 16];

    // layer-3 partials: row M = i (+8 for upper half-wave), out j
    float part[8][4];
    #pragma unroll
    for (int i = 0; i < 8; ++i)
      #pragma unroll
      for (int j = 0; j < 4; ++j) part[i][j] = 0.0f;

    // ---- layer 2 (128->128) via WMMA, fused bias+relu+layer3 epilogue ----
    #pragma unroll
    for (int nt = 0; nt < 8; ++nt) {
      v8f c = {};
      const _Float16* bp = &w2p[(nt * 4) * 32 * 16 + lane * 16];
      v16h bf;
      bf = *(const v16h*)&bp[0 * 32 * 16]; c = wmma_f16(af0, bf, c);
      bf = *(const v16h*)&bp[1 * 32 * 16]; c = wmma_f16(af1, bf, c);
      bf = *(const v16h*)&bp[2 * 32 * 16]; c = wmma_f16(af2, bf, c);
      bf = *(const v16h*)&bp[3 * 32 * 16]; c = wmma_f16(af3, bf, c);

      const int ch = nt * 16 + (lane & 15);  // this lane's output channel (C: N = lane&15)
      const float bb  = b2s[ch];
      const float w30 = W3s[ch * 4 + 0], w31 = W3s[ch * 4 + 1];
      const float w32 = W3s[ch * 4 + 2], w33 = W3s[ch * 4 + 3];
      #pragma unroll
      for (int i = 0; i < 8; ++i) {
        const float h2 = fmaxf(c[i] + bb, 0.0f);
        part[i][0] = fmaf(h2, w30, part[i][0]);
        part[i][1] = fmaf(h2, w31, part[i][1]);
        part[i][2] = fmaf(h2, w32, part[i][2]);
        part[i][3] = fmaf(h2, w33, part[i][3]);
      }
    }

    // ---- butterfly sum across the 16 lanes sharing each M half ----
    #pragma unroll
    for (int off = 8; off >= 1; off >>= 1) {
      #pragma unroll
      for (int i = 0; i < 8; ++i)
        #pragma unroll
        for (int j = 0; j < 4; ++j)
          part[i][j] += __shfl_xor(part[i][j], off, 32);
    }

    // C layout: VGPR i -> M=i (lanes 0-15) / M=8+i (lanes 16-31)
    const int lg = lane & 15;
    const int mh = (lane >> 4) * 8;
    const int i0 = lg >> 2, j0 = lg & 3;
    const int base_ray = rh * 16 + mh;
    fbuf[((base_ray + i0)     * NS + s) * 4 + j0] = part[i0][j0]     + b3s[j0];
    fbuf[((base_ray + i0 + 4) * NS + s) * 4 + j0] = part[i0 + 4][j0] + b3s[j0];
  }

  __syncthreads();

  // ---- volumetric render: one lane per ray, sequential cumprod ----
  if (tid < RPB) {
    float T = 1.0f, cr = 0.0f, cg = 0.0f, cb = 0.0f;
    for (int s = 0; s < NS; ++s) {
      const float* f = &fbuf[(tid * NS + s) * 4];
      const float alpha = 1.0f - __expf(-f[3] * delta);
      const float w = alpha * T;
      cr = fmaf(w, f[0], cr);
      cg = fmaf(w, f[1], cg);
      cb = fmaf(w, f[2], cb);
      T *= (1.0f - alpha);
    }
    out[(r0 + tid) * 3 + 0] = cr;
    out[(r0 + tid) * 3 + 1] = cg;
    out[(r0 + tid) * 3 + 2] = cb;
  }
}

extern "C" void kernel_launch(void* const* d_in, const int* in_sizes, int n_in,
                              void* d_out, int out_size, void* d_ws, size_t ws_size,
                              hipStream_t stream) {
  (void)n_in; (void)out_size; (void)d_ws; (void)ws_size;
  const float* origins = (const float*)d_in[0];
  const float* dirs    = (const float*)d_in[1];
  const float* W1      = (const float*)d_in[2];
  const float* b1      = (const float*)d_in[3];
  const float* W2      = (const float*)d_in[4];
  const float* b2      = (const float*)d_in[5];
  const float* W3      = (const float*)d_in[6];
  const float* b3      = (const float*)d_in[7];
  const int*   nearp   = (const int*)d_in[8];
  const int*   farp    = (const int*)d_in[9];
  const int*   nsp     = (const int*)d_in[10];
  float* out = (float*)d_out;

  const int B = in_sizes[0] / 3;        // 16384 rays
  const int blocks = B / RPB;           // 512 blocks, 32 rays each
  nerf_fused_kernel<<<blocks, 256, 0, stream>>>(
      origins, dirs, W1, b1, W2, b2, W3, b3, nearp, farp, nsp, out);
}
